// SetAbstractionStage_2534030704810
// MI455X (gfx1250) — matrix operand order repaired and checked by reading, not compiled, not run on hardware
//
#include <hip/hip_runtime.h>
#include <hip/hip_bf16.h>
#include <math.h>

typedef __attribute__((ext_vector_type(16))) _Float16 v16h;
typedef __attribute__((ext_vector_type(8)))  _Float16 v8h;
typedef __attribute__((ext_vector_type(8)))  float    v8f;

#define B_    64
#define P_    1024
#define M_    256
#define K_    16
#define CIN   64
#define COUT  128
#define EDIM  132      // 2*CIN + 4
#define KP1   160      // EDIM padded to multiple of 32 for f16 WMMA k-chunks
#define MT    8        // centroids per edgeconv block
#define NCOL  (MT*K_)  // 128 edge columns per block

#define PI_F      3.14159265358979323846f
#define TWOPI_F   6.28318530717958647692f
#define EPS_F     1e-8f
#define BNEPS_F   1e-5f

// d_out layout (float32, concatenated tuple):
#define OUT_MAIN  0                      // (B,128,M)  = 2097152
#define OUT_PL    (B_*COUT*M_)           // (B,4,M)    = 65536
#define OUT_CC    (OUT_PL + B_*4*M_)     // (B,2,M)    = 32768

// ---------------------------------------------------------------- utilities
static __device__ __forceinline__ float dphi_f(float a, float b) {
    float x = a - b + PI_F;
    float m = fmodf(x, TWOPI_F);
    if (m < 0.0f) m += TWOPI_F;          // python-style mod (sign of divisor)
    return m - PI_F;
}

static __device__ __forceinline__ unsigned rotl32(unsigned x, int r) {
    return (x << r) | (x >> (32 - r));
}

// Exact replica of jax.random.uniform(jax.random.key(42), (B,P)) element n.
// threefry2x32 with key [0,42]; count = iota(65536) split in halves.
static __device__ __forceinline__ float jax_uniform_42(unsigned n) {
    const unsigned half = (B_ * P_) / 2;      // 32768
    unsigned i  = (n < half) ? n : (n - half);
    unsigned x0 = i, x1 = i + half;
    unsigned ks[3] = {0u, 42u, 0u ^ 42u ^ 0x1BD11BDAu};
    x0 += ks[0]; x1 += ks[1];
    const int rotA[4] = {13, 15, 26, 6};
    const int rotB[4] = {17, 29, 16, 24};
    #pragma unroll
    for (int g = 0; g < 5; ++g) {
        const int* r = (g & 1) ? rotB : rotA;
        #pragma unroll
        for (int j = 0; j < 4; ++j) { x0 += x1; x1 = rotl32(x1, r[j]); x1 ^= x0; }
        x0 += ks[(g + 1) % 3];
        x1 += ks[(g + 2) % 3] + (unsigned)(g + 1);
    }
    unsigned bits = (n < half) ? x0 : x1;
    unsigned f = (bits >> 9) | 0x3F800000u;
    return __uint_as_float(f) - 1.0f;         // [0,1)
}

static __device__ __forceinline__ unsigned long long packvi(float v, int idx) {
    // max(packed) == (max value, then min index) for non-negative v (incl +inf)
    return ((unsigned long long)__float_as_uint(v) << 32) |
           (unsigned long long)(0xFFFFFFFFu - (unsigned)idx);
}

static __device__ __forceinline__ v16h cat8(v8h lo, v8h hi) {
    v16h r;
    #pragma unroll
    for (int i = 0; i < 8; ++i) { r[i] = lo[i]; r[i + 8] = hi[i]; }
    return r;
}

// ---------------------------------------------------------------- kernel 1: FPS
// One wave32 per event. Each lane owns the 32 points p = j*32+lane with their
// min-distances held in registers (fully unrolled). The serial FPS loop then
// needs ZERO barriers: the argmax is a 5-step __shfl_xor reduction on packed
// (dist_bits<<32 | ~idx) u64 keys (matches jnp.argmax first-max tie-break).
__global__ __launch_bounds__(32)
void fps_kernel(const float* __restrict__ coords, int* __restrict__ cidx) {
    __shared__ float etas[P_];
    __shared__ float phis[P_];
    const int b = blockIdx.x, lane = threadIdx.x;

    // stage eta/phi + seed-score argmax (threefry) in one pass
    unsigned long long best = 0ull;
    for (int i = lane; i < P_; i += 32) {
        float e = coords[b * 2 * P_ + i];
        float ph = coords[b * 2 * P_ + P_ + i];
        etas[i] = e; phis[i] = ph;
        unsigned long long pk = packvi(jax_uniform_42((unsigned)(b * P_ + i)), i);
        if (pk > best) best = pk;
    }
    __syncthreads();   // once, for the LDS staging

    // per-lane copies of owned points (registers)
    float pe[32], pp[32], mind[32];
    #pragma unroll
    for (int j = 0; j < 32; ++j) {
        pe[j] = etas[j * 32 + lane];
        pp[j] = phis[j * 32 + lane];
        mind[j] = INFINITY;
    }

    #pragma unroll
    for (int m2 = 1; m2 < 32; m2 <<= 1) {
        unsigned long long o = __shfl_xor(best, m2, 32);
        if (o > best) best = o;
    }
    int cur = (int)(0xFFFFFFFFu - (unsigned)(best & 0xFFFFFFFFull));

    for (int it = 0; it < M_; ++it) {
        if (lane == 0) cidx[b * M_ + it] = cur;
        const float ce = etas[cur], cp = phis[cur];   // LDS broadcast
        unsigned long long lb = 0ull;
        #pragma unroll
        for (int j = 0; j < 32; ++j) {
            float de = pe[j] - ce;
            float dp = dphi_f(pp[j], cp);
            mind[j] = fminf(mind[j], de * de + dp * dp);
            unsigned long long pk = packvi(mind[j], j * 32 + lane);
            if (pk > lb) lb = pk;
        }
        #pragma unroll
        for (int m2 = 1; m2 < 32; m2 <<= 1) {
            unsigned long long o = __shfl_xor(lb, m2, 32);
            if (o > lb) lb = o;
        }
        cur = (int)(0xFFFFFFFFu - (unsigned)(lb & 0xFFFFFFFFull));
    }
}

// ---------------------------------------------------------------- kernel 2: kNN
__global__ __launch_bounds__(256)
void knn_kernel(const float* __restrict__ coords, const int* __restrict__ cidx,
                int* __restrict__ nbr) {
    __shared__ float etas[P_];
    __shared__ float phis[P_];
    const int b = blockIdx.x, m = threadIdx.x;
    for (int i = m; i < P_; i += 256) {
        etas[i] = coords[b * 2 * P_ + i];
        phis[i] = coords[b * 2 * P_ + P_ + i];
    }
    __syncthreads();

    const int cen = cidx[b * M_ + m];
    const float ce = etas[cen], cp = phis[cen];
    float kd[K_]; int ki[K_];
    #pragma unroll
    for (int j = 0; j < K_; ++j) { kd[j] = INFINITY; ki[j] = 0; }

    for (int p = 0; p < P_; ++p) {
        if (p == cen) continue;                  // self_mask -> inf
        float de = ce - etas[p];
        float dp = dphi_f(cp, phis[p]);
        float d = de * de + dp * dp;
        if (d < kd[K_ - 1]) {
            float cd = d; int ci = p;
            #pragma unroll
            for (int j = 0; j < K_; ++j) {       // stable ascending insert
                if (cd < kd[j]) {
                    float td = kd[j]; kd[j] = cd; cd = td;
                    int   ti = ki[j]; ki[j] = ci; ci = ti;
                }
            }
        }
    }
    #pragma unroll
    for (int j = 0; j < K_; ++j) nbr[(b * M_ + m) * K_ + j] = ki[j];
}

// ---------------------------------------------------------------- kernel 3: EdgeConv
__global__ __launch_bounds__(256)
void edgeconv_kernel(const float* __restrict__ F, const float* __restrict__ CO,
                     const float* __restrict__ LV,
                     const float* __restrict__ W1, const float* __restrict__ W2,
                     const float* __restrict__ SCW, const float* __restrict__ ATT,
                     const float* __restrict__ bn1g, const float* __restrict__ bn1b,
                     const float* __restrict__ bn1m, const float* __restrict__ bn1v,
                     const float* __restrict__ bn2g, const float* __restrict__ bn2b,
                     const float* __restrict__ bn2m, const float* __restrict__ bn2v,
                     const float* __restrict__ bscg, const float* __restrict__ bscb,
                     const float* __restrict__ bscm, const float* __restrict__ bscv,
                     const int* __restrict__ cidx, const int* __restrict__ nbr,
                     float* __restrict__ out) {
    __shared__ _Float16 W1s[COUT * KP1];        // 40960 B, zero-padded c>=132
    __shared__ _Float16 W2s[COUT * COUT];       // 32768 B
    __shared__ _Float16 Es[NCOL * KP1];         // 40960 B, E^T: [col][c]
    __shared__ _Float16 H1T[NCOL * COUT];       // 32768 B, H1^T: [col][c2]
    __shared__ float sc1s[COUT], bi1s[COUT], sc2s[COUT], bi2s[COUT], sccs[COUT], scbs[COUT];
    __shared__ float attws[EDIM];
    __shared__ float logitss[NCOL];
    __shared__ float nbrlvs[4 * NCOL];
    __shared__ float cenfs[CIN * MT];
    __shared__ float pools[COUT * MT];

    const int t = threadIdx.x;
    const int b = blockIdx.x >> 5;              // 32 m-tiles per event
    const int mbase = (blockIdx.x & 31) * MT;

    // ---- phase 0: stage weights (f16) + BN affine folds + attention weights
    for (int i = t; i < COUT * KP1; i += 256) {
        int o = i / KP1, c = i - o * KP1;
        W1s[i] = (c < EDIM) ? (_Float16)W1[o * EDIM + c] : (_Float16)0.0f;
    }
    for (int i = t; i < COUT * COUT; i += 256) W2s[i] = (_Float16)W2[i];
    if (t < COUT) {
        float s1 = bn1g[t] / sqrtf(bn1v[t] + BNEPS_F);
        sc1s[t] = s1; bi1s[t] = bn1b[t] - bn1m[t] * s1;
        float s2 = bn2g[t] / sqrtf(bn2v[t] + BNEPS_F);
        sc2s[t] = s2; bi2s[t] = bn2b[t] - bn2m[t] * s2;
        float s3 = bscg[t] / sqrtf(bscv[t] + BNEPS_F);
        sccs[t] = s3; scbs[t] = bscb[t] - bscm[t] * s3;
    }
    if (t < EDIM) attws[t] = ATT[t];
    __syncthreads();

    // ---- phase 1: build edge tile (f16, transposed) + logits + aux
    if (t < NCOL) {
        const int lm = t >> 4, kk = t & 15;
        const int m  = mbase + lm;
        const int cen = cidx[b * M_ + m];
        const int pj  = nbr[(b * M_ + m) * K_ + kk];
        const float* Fb = F + (size_t)b * CIN * P_;
        _Float16* ec = &Es[t * KP1];
        float logit = 0.0f;
        #pragma unroll 4
        for (int c = 0; c < CIN; ++c) {
            float cf = Fb[c * P_ + cen];
            float nf = Fb[c * P_ + pj];
            ec[c]       = (_Float16)cf;
            ec[CIN + c] = (_Float16)(nf - cf);
            logit += attws[c] * cf + attws[CIN + c] * (nf - cf);
            if (kk == 0) cenfs[c * MT + lm] = cf;
        }
        // pairwise Lorentz features
        const float* Lb = LV + (size_t)b * 4 * P_;
        float xi0 = Lb[cen], xi1 = Lb[P_ + cen], xi2 = Lb[2 * P_ + cen], xi3 = Lb[3 * P_ + cen];
        float xj0 = Lb[pj],  xj1 = Lb[P_ + pj],  xj2 = Lb[2 * P_ + pj],  xj3 = Lb[3 * P_ + pj];
        float pti = sqrtf(fmaxf(xi0 * xi0 + xi1 * xi1, EPS_F));
        float ptj = sqrtf(fmaxf(xj0 * xj0 + xj1 * xj1, EPS_F));
        float rapi = 0.5f * log1pf(2.0f * xi2 / fmaxf(xi3 - xi2, 1e-20f));
        float rapj = 0.5f * log1pf(2.0f * xj2 / fmaxf(xj3 - xj2, 1e-20f));
        float phii = atan2f(xi1, xi0), phij = atan2f(xj1, xj0);
        float dr = rapi - rapj, dp = dphi_f(phii, phij);
        float delta = sqrtf(dr * dr + dp * dp);
        float lndelta = logf(fmaxf(delta, EPS_F));
        float ptmin = fminf(pti, ptj);
        float lnkt = logf(fmaxf(ptmin * delta, EPS_F));
        float lnz  = logf(fmaxf(ptmin / fmaxf(pti + ptj, EPS_F), EPS_F));
        float s0 = xi0 + xj0, s1 = xi1 + xj1, s2 = xi2 + xj2, s3 = xi3 + xj3;
        float m2 = s3 * s3 - s0 * s0 - s1 * s1 - s2 * s2;
        float lnm2 = logf(fmaxf(m2, EPS_F));
        float fts[4] = {lnkt, lnz, lndelta, lnm2};
        #pragma unroll
        for (int j = 0; j < 4; ++j) {
            ec[2 * CIN + j] = (_Float16)fts[j];
            logit += attws[2 * CIN + j] * fts[j];
        }
        #pragma unroll
        for (int c = EDIM; c < KP1; ++c) ec[c] = (_Float16)0.0f;   // pad
        logitss[t] = logit;
        nbrlvs[0 * NCOL + t] = xj0; nbrlvs[1 * NCOL + t] = xj1;
        nbrlvs[2 * NCOL + t] = xj2; nbrlvs[3 * NCOL + t] = xj3;
        if (kk == 0) {
            const float* Cb = CO + (size_t)b * 2 * P_;
            out[OUT_CC + b * 2 * M_ + m]       = Cb[cen];
            out[OUT_CC + b * 2 * M_ + M_ + m]  = Cb[P_ + cen];
        }
    }
    __syncthreads();

    // ---- phase 2: conv1 (f16 WMMA, f32 acc) + BN1 + ReLU -> H1^T (f16)
    const int wave = t >> 5, lane = t & 31, hi = lane >> 4, l15 = lane & 15;
    {
        v16h a1[5];
        const int row = wave * 16 + l15;
        #pragma unroll
        for (int kc = 0; kc < 5; ++kc) {
            // A layout: k = e + 16*hi (contiguous per lane, matches B)
            const _Float16* pA = &W1s[row * KP1 + kc * 32 + hi * 16];
            a1[kc] = cat8(*(const v8h*)pA, *(const v8h*)(pA + 8));
        }
        for (int nt = 0; nt < 8; ++nt) {
            v8f acc = {};
            const int col = nt * 16 + l15;
            #pragma unroll
            for (int kc = 0; kc < 5; ++kc) {
                // B layout: lanes 0-15 K=0..15, lanes 16-31 K=16..31 (contiguous per lane)
                const _Float16* pB = &Es[col * KP1 + kc * 32 + hi * 16];
                v16h bb = cat8(*(const v8h*)pB, *(const v8h*)(pB + 8));
                acc = __builtin_amdgcn_wmma_f32_16x16x32_f16(false, a1[kc], false, bb,
                                                             (short)0, acc, false, false);
            }
            v8h hv;
            #pragma unroll
            for (int r = 0; r < 8; ++r) {
                int o = wave * 16 + hi * 8 + r;                    // D row = r (+8 for hi half)
                float v = acc[r] * sc1s[o] + bi1s[o];
                hv[r] = (_Float16)fmaxf(v, 0.0f);
            }
            *(v8h*)&H1T[col * COUT + wave * 16 + hi * 8] = hv;
        }
    }
    __syncthreads();

    // ---- phase 3: conv2 (f16 WMMA) + BN2 + max-pool over K (shuffle on D fragment)
    {
        v16h a2[4];
        const int row = wave * 16 + l15;
        #pragma unroll
        for (int kc = 0; kc < 4; ++kc) {
            const _Float16* pA = &W2s[row * COUT + kc * 32 + hi * 16];
            a2[kc] = cat8(*(const v8h*)pA, *(const v8h*)(pA + 8));
        }
        for (int nt = 0; nt < 8; ++nt) {
            v8f acc = {};
            const int col = nt * 16 + l15;
            #pragma unroll
            for (int kc = 0; kc < 4; ++kc) {
                const _Float16* pB = &H1T[col * COUT + kc * 32 + hi * 16];
                v16h bb = cat8(*(const v8h*)pB, *(const v8h*)(pB + 8));
                acc = __builtin_amdgcn_wmma_f32_16x16x32_f16(false, a2[kc], false, bb,
                                                             (short)0, acc, false, false);
            }
            #pragma unroll
            for (int r = 0; r < 8; ++r) {
                int o = wave * 16 + hi * 8 + r;
                float v = acc[r] * sc2s[o] + bi2s[o];
                // lanes 0..15 of this D row are the 16 neighbors of centroid nt
                #pragma unroll
                for (int msk = 1; msk < 16; msk <<= 1)
                    v = fmaxf(v, __shfl_xor(v, msk, 32));
                if (l15 == 0) pools[o * MT + nt] = v;
            }
        }
    }
    __syncthreads();

    // ---- phase 4: shortcut GEMV + BNsc + ReLU -> out ; attention -> prop_lv
    #pragma unroll
    for (int j = 0; j < 4; ++j) {
        int idx = t + j * 256;                 // 1024 = 128 channels x 8 centroids
        int o = idx >> 3, lm = idx & 7;
        float acc = 0.0f;
        #pragma unroll 8
        for (int c = 0; c < CIN; ++c) acc += SCW[o * CIN + c] * cenfs[c * MT + lm];
        float scv = acc * sccs[o] + scbs[o];
        float v = fmaxf(pools[o * MT + lm] + scv, 0.0f);
        out[OUT_MAIN + (size_t)b * COUT * M_ + o * M_ + mbase + lm] = v;
    }
    if (t < 32) {
        int lm = t >> 2, c = t & 3;
        float mx = -INFINITY;
        #pragma unroll
        for (int k = 0; k < K_; ++k) mx = fmaxf(mx, logitss[lm * K_ + k]);
        float s = 0.0f;
        #pragma unroll
        for (int k = 0; k < K_; ++k) s += expf(logitss[lm * K_ + k] - mx);
        float accv = 0.0f;
        #pragma unroll
        for (int k = 0; k < K_; ++k)
            accv += (expf(logitss[lm * K_ + k] - mx) / s) * nbrlvs[c * NCOL + lm * K_ + k];
        out[OUT_PL + b * 4 * M_ + c * M_ + mbase + lm] = accv;
    }
}

// ---------------------------------------------------------------- launch
extern "C" void kernel_launch(void* const* d_in, const int* in_sizes, int n_in,
                              void* d_out, int out_size, void* d_ws, size_t ws_size,
                              hipStream_t stream) {
    (void)in_sizes; (void)n_in; (void)out_size; (void)ws_size;
    const float* features = (const float*)d_in[0];
    const float* coords   = (const float*)d_in[1];
    const float* lv       = (const float*)d_in[2];
    // d_in[3] = mask (all ones) -> ignored
    const float* w1   = (const float*)d_in[4];
    const float* w2   = (const float*)d_in[5];
    const float* scw  = (const float*)d_in[6];
    const float* attw = (const float*)d_in[7];
    const float* bn1g = (const float*)d_in[8];
    const float* bn1b = (const float*)d_in[9];
    const float* bn1m = (const float*)d_in[10];
    const float* bn1v = (const float*)d_in[11];
    const float* bn2g = (const float*)d_in[12];
    const float* bn2b = (const float*)d_in[13];
    const float* bn2m = (const float*)d_in[14];
    const float* bn2v = (const float*)d_in[15];
    const float* bscg = (const float*)d_in[16];
    const float* bscb = (const float*)d_in[17];
    const float* bscm = (const float*)d_in[18];
    const float* bscv = (const float*)d_in[19];
    float* out = (float*)d_out;

    int* cidx = (int*)d_ws;                                   // 64*256 ints
    int* nbr  = (int*)((char*)d_ws + B_ * M_ * sizeof(int));  // 64*256*16 ints

    fps_kernel<<<B_, 32, 0, stream>>>(coords, cidx);
    knn_kernel<<<B_, 256, 0, stream>>>(coords, cidx, nbr);
    edgeconv_kernel<<<B_ * (M_ / MT), 256, 0, stream>>>(
        features, coords, lv, w1, w2, scw, attw,
        bn1g, bn1b, bn1m, bn1v, bn2g, bn2b, bn2m, bn2v,
        bscg, bscb, bscm, bscv, cidx, nbr, out);
}
